// GCN_25074019074551
// MI455X (gfx1250) — matrix operand order repaired and checked by reading, not compiled, not run on hardware
//
#include <hip/hip_runtime.h>
#include <hip/hip_bf16.h>

// ---------------------------------------------------------------------------
// GIN forward for MI455X (gfx1250, wave32).
//  - neighbor aggregation: f32 L2 atomics (everything L2-resident, 192MB)
//  - node features between layers kept in bf16 (halves gather traffic)
//  - both per-layer GEMMs via v_wmma_f32_16x16x32_bf16, one wave per 16-row tile
// ---------------------------------------------------------------------------

typedef __attribute__((ext_vector_type(16))) __bf16 v16bf;
typedef __attribute__((ext_vector_type(8)))  __bf16 v8bf;
typedef __attribute__((ext_vector_type(8)))  float  v8f;

#define N_NODES 50000
#define N_EDGES 800000
#define N_GRAPHS 512
#define F_IN 128
#define HID 64
#define N_CLS 10

// ---------------------------------------------------------------- utilities
__global__ void zero_f32_kernel(float* __restrict__ p, int n) {
    int i = blockIdx.x * blockDim.x + threadIdx.x;
    if (i < n) p[i] = 0.0f;
}

// Repack a row-major f32 weight [K][64] into the WMMA B-matrix register
// layout for v_wmma_*_16x16x32_bf16:
//   out[ ((kt*4 + nt)*32 + lane)*16 + e ]  with
//   col = nt*16 + (lane&15),  K = kt*32 + (e>>3)*16 + (lane>>4)*8 + (e&7)
// so each lane of a wave loads its 16 bf16 B elements as one contiguous,
// 32-byte-aligned chunk.
__global__ void pack_w_kernel(const float* __restrict__ W,
                              __bf16* __restrict__ out, int K) {
    int i = blockIdx.x * blockDim.x + threadIdx.x;
    if (i >= K * 64) return;
    int e    = i & 15;
    int lane = (i >> 4) & 31;
    int t    = i >> 9;          // kt*4 + nt
    int nt   = t & 3;
    int kt   = t >> 2;
    int col  = nt * 16 + (lane & 15);
    int kk   = kt * 32 + (e >> 3) * 16 + (lane >> 4) * 8 + (e & 7);
    out[i] = (__bf16)W[kk * 64 + col];
}

// --------------------------------------------------- edge gather/scatter-add
// One thread per (edge, feature). Consecutive threads cover consecutive
// features of one edge -> coalesced row gather, coalesced atomic row scatter.
template <typename T>
__global__ void scatter_add_kernel(const T* __restrict__ h,
                                   const int* __restrict__ src,
                                   const int* __restrict__ dst,
                                   float* __restrict__ agg,
                                   int total, int logF) {
    int i = blockIdx.x * blockDim.x + threadIdx.x;
    if (i >= total) return;
    int e = i >> logF;
    int f = i & ((1 << logF) - 1);
    int s = src[e];
    int d = dst[e];
    float v = (float)h[(s << logF) + f];
    atomicAdd(&agg[(d << logF) + f], v);
}

// ------------------------------------------------------------- WMMA helpers
// A-matrix 16x32 bf16 register layout (ISA 7.12.2):
//   lanes 0-15 : row M = lane,    elements 0..7 -> K 0..7,  8..15 -> K 16..23
//   lanes 16-31: row M = lane-16, elements 0..7 -> K 8..15, 8..15 -> K 24..31
__device__ inline v16bf load_a_from_lds(const __bf16* __restrict__ lds,
                                        int ldk, int koff, int lane) {
    int row = lane & 15;
    int hi  = lane >> 4;
    const __bf16* base = lds + row * ldk + koff + hi * 8;
    v8bf a_lo = *(const v8bf*)(base);        // K = koff + hi*8 .. +7
    v8bf a_hi = *(const v8bf*)(base + 16);   // K = koff + 16 + hi*8 .. +7
    return __builtin_shufflevector(a_lo, a_hi,
                                   0, 1, 2, 3, 4, 5, 6, 7,
                                   8, 9, 10, 11, 12, 13, 14, 15);
}

// C/D layout: VGPR r holds (M = r + 8*(lane>>4), N = lane&15).

// -------------------------------------------------------------- GIN layer 0
// z = x + agg (f32), t = relu(z@W1+b1), h = relu(t@W2+b2)
// one wave (32 threads) per 16-node tile;  50000/16 = 3125 tiles exactly.
__global__ __launch_bounds__(32)
void gin_layer0_kernel(const float* __restrict__ x,
                       const float* __restrict__ agg,
                       const __bf16* __restrict__ w1p, const float* __restrict__ b1,
                       const __bf16* __restrict__ w2p, const float* __restrict__ b2,
                       __bf16* __restrict__ hout, float* __restrict__ hcat) {
    __shared__ __align__(32) __bf16 ldsZ[16 * F_IN];
    __shared__ __align__(32) __bf16 ldsT[16 * HID];
    const int lane = threadIdx.x;
    const int rowbase = blockIdx.x * 16;
    const int lo = lane & 15, hi = lane >> 4;

    // stage z tile (bf16) into LDS
    for (int i = lane; i < 16 * F_IN; i += 32) {
        int r = i >> 7, c = i & (F_IN - 1);
        int g = rowbase + r;
        ldsZ[i] = (__bf16)(x[g * F_IN + c] + agg[g * F_IN + c]);
    }
    __syncthreads();

    // GEMM 1: [16 x 128] @ [128 x 64]
    v8f acc[4];
#pragma unroll
    for (int nt = 0; nt < 4; ++nt)
#pragma unroll
        for (int r = 0; r < 8; ++r) acc[nt][r] = 0.0f;

#pragma unroll
    for (int kt = 0; kt < F_IN / 32; ++kt) {
        v16bf a = load_a_from_lds(ldsZ, F_IN, kt * 32, lane);
#pragma unroll
        for (int nt = 0; nt < 4; ++nt) {
            v16bf b = *(const v16bf*)&w1p[((kt * 4 + nt) * 32 + lane) * 16];
            acc[nt] = __builtin_amdgcn_wmma_f32_16x16x32_bf16(
                false, a, false, b, (short)0, acc[nt], false, false);
        }
    }

    // bias + relu -> LDS (bf16) for second GEMM
#pragma unroll
    for (int nt = 0; nt < 4; ++nt) {
        float bb = b1[nt * 16 + lo];
#pragma unroll
        for (int r = 0; r < 8; ++r) {
            float v = acc[nt][r] + bb;
            v = v > 0.0f ? v : 0.0f;
            ldsT[(r + 8 * hi) * HID + nt * 16 + lo] = (__bf16)v;
        }
    }
    __syncthreads();

    // GEMM 2: [16 x 64] @ [64 x 64]
    v8f o[4];
#pragma unroll
    for (int nt = 0; nt < 4; ++nt)
#pragma unroll
        for (int r = 0; r < 8; ++r) o[nt][r] = 0.0f;

#pragma unroll
    for (int kt = 0; kt < HID / 32; ++kt) {
        v16bf a = load_a_from_lds(ldsT, HID, kt * 32, lane);
#pragma unroll
        for (int nt = 0; nt < 4; ++nt) {
            v16bf b = *(const v16bf*)&w2p[((kt * 4 + nt) * 32 + lane) * 16];
            o[nt] = __builtin_amdgcn_wmma_f32_16x16x32_bf16(
                false, a, false, b, (short)0, o[nt], false, false);
        }
    }

    // bias + outer relu, write bf16 h and f32 hcat stripe (cols 0..63)
#pragma unroll
    for (int nt = 0; nt < 4; ++nt) {
        float bb = b2[nt * 16 + lo];
        int c = nt * 16 + lo;
#pragma unroll
        for (int r = 0; r < 8; ++r) {
            float v = o[nt][r] + bb;
            v = v > 0.0f ? v : 0.0f;
            int g = rowbase + r + 8 * hi;
            hout[g * HID + c] = (__bf16)v;
            hcat[g * (4 * HID) + c] = v;
        }
    }
}

// ---------------------------------------------------------- GIN layers 1..3
// identical but K_in = 64, input is bf16 h; in-place safe (each wave only
// touches its own 16 rows of h, neighbor info already captured in agg).
__global__ __launch_bounds__(32)
void gin_layer_mid_kernel(const __bf16* __restrict__ hin,
                          const float* __restrict__ agg,
                          const __bf16* __restrict__ w1p, const float* __restrict__ b1,
                          const __bf16* __restrict__ w2p, const float* __restrict__ b2,
                          __bf16* __restrict__ hout, float* __restrict__ hcat,
                          int layer_off) {
    __shared__ __align__(32) __bf16 ldsZ[16 * HID];
    __shared__ __align__(32) __bf16 ldsT[16 * HID];
    const int lane = threadIdx.x;
    const int rowbase = blockIdx.x * 16;
    const int lo = lane & 15, hi = lane >> 4;

    for (int i = lane; i < 16 * HID; i += 32) {
        int r = i >> 6, c = i & (HID - 1);
        int g = rowbase + r;
        ldsZ[i] = (__bf16)((float)hin[g * HID + c] + agg[g * HID + c]);
    }
    __syncthreads();

    v8f acc[4];
#pragma unroll
    for (int nt = 0; nt < 4; ++nt)
#pragma unroll
        for (int r = 0; r < 8; ++r) acc[nt][r] = 0.0f;

#pragma unroll
    for (int kt = 0; kt < HID / 32; ++kt) {
        v16bf a = load_a_from_lds(ldsZ, HID, kt * 32, lane);
#pragma unroll
        for (int nt = 0; nt < 4; ++nt) {
            v16bf b = *(const v16bf*)&w1p[((kt * 4 + nt) * 32 + lane) * 16];
            acc[nt] = __builtin_amdgcn_wmma_f32_16x16x32_bf16(
                false, a, false, b, (short)0, acc[nt], false, false);
        }
    }

#pragma unroll
    for (int nt = 0; nt < 4; ++nt) {
        float bb = b1[nt * 16 + lo];
#pragma unroll
        for (int r = 0; r < 8; ++r) {
            float v = acc[nt][r] + bb;
            v = v > 0.0f ? v : 0.0f;
            ldsT[(r + 8 * hi) * HID + nt * 16 + lo] = (__bf16)v;
        }
    }
    __syncthreads();

    v8f o[4];
#pragma unroll
    for (int nt = 0; nt < 4; ++nt)
#pragma unroll
        for (int r = 0; r < 8; ++r) o[nt][r] = 0.0f;

#pragma unroll
    for (int kt = 0; kt < HID / 32; ++kt) {
        v16bf a = load_a_from_lds(ldsT, HID, kt * 32, lane);
#pragma unroll
        for (int nt = 0; nt < 4; ++nt) {
            v16bf b = *(const v16bf*)&w2p[((kt * 4 + nt) * 32 + lane) * 16];
            o[nt] = __builtin_amdgcn_wmma_f32_16x16x32_bf16(
                false, a, false, b, (short)0, o[nt], false, false);
        }
    }

#pragma unroll
    for (int nt = 0; nt < 4; ++nt) {
        float bb = b2[nt * 16 + lo];
        int c = nt * 16 + lo;
#pragma unroll
        for (int r = 0; r < 8; ++r) {
            float v = o[nt][r] + bb;
            v = v > 0.0f ? v : 0.0f;
            int g = rowbase + r + 8 * hi;
            hout[g * HID + c] = (__bf16)v;
            hcat[g * (4 * HID) + layer_off + c] = v;
        }
    }
}

// ----------------------------------------------------------------- pooling
__global__ void pool_sum_kernel(const float* __restrict__ hcat,
                                const int* __restrict__ batch,
                                float* __restrict__ pooled, int total) {
    int i = blockIdx.x * blockDim.x + threadIdx.x;
    if (i >= total) return;
    int node = i >> 8, c = i & 255;
    atomicAdd(&pooled[batch[node] * 256 + c], hcat[i]);
}

__global__ void pool_cnt_kernel(const int* __restrict__ batch,
                                float* __restrict__ cnt) {
    int i = blockIdx.x * blockDim.x + threadIdx.x;
    if (i < N_NODES) atomicAdd(&cnt[batch[i]], 1.0f);
}

// -------------------------------------------------------- classifier head
__global__ __launch_bounds__(32)
void head_kernel(const float* __restrict__ pooled, const float* __restrict__ cnt,
                 const float* __restrict__ fcw, const float* __restrict__ fcb,
                 float* __restrict__ out) {
    __shared__ float logits[16];
    int g = blockIdx.x, j = threadIdx.x;
    float inv = 1.0f / fmaxf(cnt[g], 1.0f);
    if (j < N_CLS) {
        float s = fcb[j];
        for (int c = 0; c < 4 * HID; ++c)
            s += pooled[g * 256 + c] * inv * fcw[c * N_CLS + j];
        logits[j] = s;
    }
    __syncthreads();
    if (j < N_CLS) {
        float m = -1e30f;
        for (int k = 0; k < N_CLS; ++k) m = fmaxf(m, logits[k]);
        float se = 0.0f;
        for (int k = 0; k < N_CLS; ++k) se += __expf(logits[k] - m);
        out[g * N_CLS + j] = (logits[j] - m) - __logf(se);
    }
}

// ------------------------------------------------------------------ launch
extern "C" void kernel_launch(void* const* d_in, const int* in_sizes, int n_in,
                              void* d_out, int out_size, void* d_ws, size_t ws_size,
                              hipStream_t stream) {
    (void)in_sizes; (void)n_in; (void)out_size; (void)ws_size;
    const float* x    = (const float*)d_in[0];
    const int*   ei   = (const int*)d_in[1];
    const int*   batch= (const int*)d_in[2];
    const float* w1_0 = (const float*)d_in[3];
    const float* b1_0 = (const float*)d_in[4];
    const float* w2_0 = (const float*)d_in[5];
    const float* b2_0 = (const float*)d_in[6];
    const float* w1_r = (const float*)d_in[7];
    const float* b1_r = (const float*)d_in[8];
    const float* w2_r = (const float*)d_in[9];
    const float* b2_r = (const float*)d_in[10];
    const float* fcw  = (const float*)d_in[11];
    const float* fcb  = (const float*)d_in[12];
    float* out = (float*)d_out;

    const int* src = ei;
    const int* dst = ei + N_EDGES;

    // workspace carve-out (all 256B aligned)
    char* ws = (char*)d_ws;
    size_t off = 0;
    auto carve = [&](size_t bytes) -> void* {
        void* p = ws + off;
        off = (off + bytes + 255) & ~(size_t)255;
        return p;
    };
    float*  agg    = (float*)carve((size_t)N_NODES * F_IN * 4);
    __bf16* h      = (__bf16*)carve((size_t)N_NODES * HID * 2);
    float*  hcat   = (float*)carve((size_t)N_NODES * 4 * HID * 4);
    float*  pooled = (float*)carve((size_t)N_GRAPHS * 256 * 4);
    float*  cnt    = (float*)carve((size_t)N_GRAPHS * 4);
    __bf16* w1_0p  = (__bf16*)carve((size_t)F_IN * HID * 2);
    __bf16* w2_0p  = (__bf16*)carve((size_t)HID * HID * 2);
    __bf16* w1_rp  = (__bf16*)carve((size_t)3 * HID * HID * 2);
    __bf16* w2_rp  = (__bf16*)carve((size_t)3 * HID * HID * 2);

    // pack weights into WMMA B layout (bf16)
    pack_w_kernel<<<(F_IN * 64 + 255) / 256, 256, 0, stream>>>(w1_0, w1_0p, F_IN);
    pack_w_kernel<<<(HID * 64 + 255) / 256, 256, 0, stream>>>(w2_0, w2_0p, HID);
    for (int r = 0; r < 3; ++r) {
        pack_w_kernel<<<(HID * 64 + 255) / 256, 256, 0, stream>>>(
            w1_r + r * HID * HID, w1_rp + r * HID * HID, HID);
        pack_w_kernel<<<(HID * 64 + 255) / 256, 256, 0, stream>>>(
            w2_r + r * HID * HID, w2_rp + r * HID * HID, HID);
    }

    // ---- layer 0 (F_IN = 128)
    {
        int n = N_NODES * F_IN;
        zero_f32_kernel<<<(n + 255) / 256, 256, 0, stream>>>(agg, n);
        int tot = N_EDGES * F_IN;
        scatter_add_kernel<float><<<(tot + 255) / 256, 256, 0, stream>>>(
            x, src, dst, agg, tot, 7);
        gin_layer0_kernel<<<N_NODES / 16, 32, 0, stream>>>(
            x, agg, w1_0p, b1_0, w2_0p, b2_0, h, hcat);
    }

    // ---- layers 1..3 (HID = 64)
    for (int r = 0; r < 3; ++r) {
        int n = N_NODES * HID;
        zero_f32_kernel<<<(n + 255) / 256, 256, 0, stream>>>(agg, n);
        int tot = N_EDGES * HID;
        scatter_add_kernel<__bf16><<<(tot + 255) / 256, 256, 0, stream>>>(
            h, src, dst, agg, tot, 6);
        gin_layer_mid_kernel<<<N_NODES / 16, 32, 0, stream>>>(
            h, agg, w1_rp + r * HID * HID, b1_r + r * HID,
            w2_rp + r * HID * HID, b2_r + r * HID, h, hcat, (r + 1) * HID);
    }

    // ---- pooling + head
    zero_f32_kernel<<<(N_GRAPHS * 256 + 255) / 256, 256, 0, stream>>>(pooled, N_GRAPHS * 256);
    zero_f32_kernel<<<(N_GRAPHS + 255) / 256, 256, 0, stream>>>(cnt, N_GRAPHS);
    {
        int tot = N_NODES * 256;
        pool_sum_kernel<<<(tot + 255) / 256, 256, 0, stream>>>(hcat, batch, pooled, tot);
        pool_cnt_kernel<<<(N_NODES + 255) / 256, 256, 0, stream>>>(batch, cnt);
    }
    head_kernel<<<N_GRAPHS, 32, 0, stream>>>(pooled, cnt, fcw, fcb, out);
}
